// LlamaLolcatsAttentionActual_13572096655482
// MI455X (gfx1250) — compile-verified, matrix operand order checked by reading, not run on hardware
//
#include <hip/hip_runtime.h>
#include <hip/hip_bf16.h>

// ---------------------------------------------------------------------------
// LlamaLolcats hybrid attention (sliding-window softmax + linear attention)
// CDNA5 / gfx1250, wave32, v_wmma_f32_16x16x32_bf16 for all matmuls.
// All WMMA operands staged K-contiguous in LDS -> ds_load_b128 fragments.
// Shapes: B=1, H=32, KVH=8, L=2048, D=128, F=64, F2=128, chunk=window=64.
// ---------------------------------------------------------------------------

#define HH   32
#define KVHH 8
#define LL   2048
#define DD   128
#define FF   64
#define F2   128
#define CHK  64
#define NCK  32   // LL / CHK

typedef __attribute__((ext_vector_type(16))) __bf16 v16bf;
typedef __attribute__((ext_vector_type(8)))  __bf16 v8bf;
typedef __attribute__((ext_vector_type(4)))  __bf16 v4bf;
typedef __attribute__((ext_vector_type(8)))  float  v8f;
typedef __attribute__((ext_vector_type(4)))  float  v4f;

__device__ __forceinline__ v8f vzero8() {
  v8f z = {0.f, 0.f, 0.f, 0.f, 0.f, 0.f, 0.f, 0.f};
  return z;
}

__device__ __forceinline__ v8bf bzero8() {
  v8bf z;
#pragma unroll
  for (int e = 0; e < 8; ++e) z[e] = (__bf16)0.0f;
  return z;
}

__device__ __forceinline__ v8f wmma_bf16(v16bf a, v16bf b, v8f c) {
  // (neg_a, A, neg_b, B, c_mod, C, reuse_a, reuse_b)
  return __builtin_amdgcn_wmma_f32_16x16x32_bf16(false, a, false, b,
                                                 (short)0, c, false, false);
}

// Fragment loader for K-contiguous storage (works for A [M][K] and for B
// staged transposed as [N][K]).  Lane l: row = mn0 + (l&15); needs
// K = k0 + 8*(l>>4) + {0..7} and + {16..23}  ->  two 16-byte LDS loads.
__device__ __forceinline__ v16bf load_kmaj_tile(const __bf16* p, int ld,
                                                int mn0, int k0) {
  const int lane = threadIdx.x & 31;
  const int m = lane & 15, half = lane >> 4;
  const __bf16* rowp = p + (size_t)(mn0 + m) * ld + k0 + 8 * half;
  union { v16bf v; v8bf h[2]; } u;
  u.h[0] = *(const v8bf*)(rowp);        // ds_load_b128
  u.h[1] = *(const v8bf*)(rowp + 16);   // ds_load_b128
  return u.v;
}

// ---------------------------------------------------------------------------
// K0: f32 -> bf16 convert, 4 elements / thread (b128 load, b64 store)
// ---------------------------------------------------------------------------
__global__ void cvt_f32_bf16_kernel(const float* __restrict__ x,
                                    __bf16* __restrict__ y, int n4) {
  int i = blockIdx.x * 256 + threadIdx.x;
  if (i < n4) {
    v4f xv = *(const v4f*)(x + (size_t)i * 4);
    v4bf o;
#pragma unroll
    for (int e = 0; e < 4; ++e) o[e] = (__bf16)xv[e];
    *(v4bf*)(y + (size_t)i * 4) = o;
  }
}

// ---------------------------------------------------------------------------
// K1: feature map  f = clip(concat(softmax(xW), softmax(-xW)), EPS)
// grid (H, NC), 256 threads. head_div = 1 for q, 4 for k (GQA repeat).
// ---------------------------------------------------------------------------
__global__ void feature_kernel(const float* __restrict__ X,   // [H/hd][L][D]
                               const float* __restrict__ W,   // [H][D][F]
                               __bf16* __restrict__ Fout,     // [H][L][F2]
                               int head_div) {
  const int h = blockIdx.x, ci = blockIdx.y;
  const int tid = threadIdx.x, wave = tid >> 5, lane = tid & 31;
  const int hm = lane >> 4, nlo = lane & 15;

  __shared__ __align__(16) __bf16 sx[CHK * 136];   // x chunk [m][d] (ld 136)
  __shared__ __align__(16) __bf16 swT[FF * 136];   // W^T [f][d]    (ld 136)
  __shared__ __align__(16) float  sz[CHK * 68];    // z [m][f]      (ld 68)

  const float* xrow = X + ((size_t)(h / head_div) * LL + (size_t)ci * CHK) * DD;
  for (int idx = tid; idx < CHK * DD / 4; idx += 256) {   // 2048
    int r = idx >> 5, c4 = (idx & 31) * 4;
    v4f xv = *(const v4f*)(xrow + (size_t)r * DD + c4);
    v4bf o;
#pragma unroll
    for (int e = 0; e < 4; ++e) o[e] = (__bf16)xv[e];
    *(v4bf*)(sx + r * 136 + c4) = o;
  }
  const float* wp = W + (size_t)h * DD * FF;
  for (int idx = tid; idx < DD * FF / 4; idx += 256) {    // 2048
    int r = idx >> 4, c4 = (idx & 15) * 4;                // r: d, c4: f
    v4f wv = *(const v4f*)(wp + (size_t)r * FF + c4);
#pragma unroll
    for (int e = 0; e < 4; ++e) swT[(c4 + e) * 136 + r] = (__bf16)wv[e];
  }
  __syncthreads();

  // z = X(64x128) @ W(128x64): 4x4 tiles, 2 tiles/wave, K-steps = 4
  const int mt = wave & 3;
#pragma unroll
  for (int t = 0; t < 2; ++t) {
    const int nt = (wave >> 2) * 2 + t;
    v8f acc = vzero8();
#pragma unroll
    for (int kk = 0; kk < 4; ++kk) {
      v16bf a = load_kmaj_tile(sx, 136, mt * 16, kk * 32);
      v16bf b = load_kmaj_tile(swT, 136, nt * 16, kk * 32);
      acc = wmma_bf16(a, b, acc);
    }
#pragma unroll
    for (int r = 0; r < 8; ++r)
      sz[(mt * 16 + 8 * hm + r) * 68 + nt * 16 + nlo] = acc[r];
  }
  __syncthreads();

  // dual softmax + clip, one row per thread (first 64 threads)
  if (tid < CHK) {
    float zr[FF];
    float mx = -3.0e38f, mn = 3.0e38f;
#pragma unroll 8
    for (int c = 0; c < FF; ++c) {
      zr[c] = sz[tid * 68 + c];
      mx = fmaxf(mx, zr[c]);
      mn = fminf(mn, zr[c]);
    }
    float sp = 0.f, sn = 0.f;
#pragma unroll 8
    for (int c = 0; c < FF; ++c) {
      sp += __expf(zr[c] - mx);
      sn += __expf(mn - zr[c]);   // softmax(-z): max(-z) = -min(z)
    }
    const float rp = 1.f / sp, rn = 1.f / sn;
    __bf16* orow = Fout + ((size_t)h * LL + (size_t)ci * CHK + tid) * F2;
#pragma unroll 8
    for (int c = 0; c < FF; ++c) {
      float fp = fmaxf(__expf(zr[c] - mx) * rp, 1e-12f);
      float fn = fmaxf(__expf(mn - zr[c]) * rn, 1e-12f);
      orow[c]      = (__bf16)fp;
      orow[FF + c] = (__bf16)fn;
    }
  }
}

// ---------------------------------------------------------------------------
// K2: per-chunk state partials  S_j = f_k^T (128x64) @ v (64x128),
//     stored TRANSPOSED as S^T [d][f];  z_j = column sums of f_k.
// grid (H, NC). Operand regions reused as the S^T bounce buffer.
// ---------------------------------------------------------------------------
__global__ void state_partial_kernel(const __bf16* __restrict__ Fk, // [H][L][F2]
                                     const __bf16* __restrict__ Vb, // [KVH][L][D]
                                     __bf16* __restrict__ Sp,       // [H][NC][D][F2] (S^T)
                                     float* __restrict__ Zp) {      // [H][NC][F2]
  const int h = blockIdx.x, ci = blockIdx.y;
  const int tid = threadIdx.x, wave = tid >> 5, lane = tid & 31;
  const int hm = lane >> 4, nlo = lane & 15;

  __shared__ __align__(16) unsigned char smem[36864];
  __bf16* sa  = (__bf16*)smem;             // f_k^T [f][n], 128x64 (ld 72)
  __bf16* sbT = (__bf16*)(smem + 18432);   // v^T   [d][n], 128x64 (ld 72)

  const __bf16* fkp = Fk + ((size_t)h * LL + (size_t)ci * CHK) * F2;
  for (int idx = tid; idx < CHK * F2 / 8; idx += 256) {   // 1024
    int n = idx >> 4, f8 = (idx & 15) * 8;
    v8bf fv = *(const v8bf*)(fkp + (size_t)n * F2 + f8);
#pragma unroll
    for (int e = 0; e < 8; ++e) sa[(f8 + e) * 72 + n] = fv[e];
  }
  const __bf16* vp = Vb + ((size_t)(h >> 2) * LL + (size_t)ci * CHK) * DD;
  for (int idx = tid; idx < CHK * DD / 8; idx += 256) {   // 1024
    int n = idx >> 4, d8 = (idx & 15) * 8;
    v8bf vv = *(const v8bf*)(vp + (size_t)n * DD + d8);
#pragma unroll
    for (int e = 0; e < 8; ++e) sbT[(d8 + e) * 72 + n] = vv[e];
  }
  __syncthreads();

  // z partial (read sa before region reuse)
  float zval = 0.f;
  if (tid < F2) {
#pragma unroll 8
    for (int n = 0; n < CHK; ++n) zval += (float)sa[tid * 72 + n];
  }

  // S = A(128x64) @ B(64x128): 8x8 tiles, wave = f-row-tile, K-steps = 2
  const int mt = wave;
  v8f acc[8];
#pragma unroll
  for (int nt = 0; nt < 8; ++nt) {
    acc[nt] = vzero8();
#pragma unroll
    for (int kk = 0; kk < 2; ++kk) {
      v16bf a = load_kmaj_tile(sa, 72, mt * 16, kk * 32);
      v16bf b = load_kmaj_tile(sbT, 72, nt * 16, kk * 32);
      acc[nt] = wmma_bf16(a, b, acc[nt]);
    }
  }
  __syncthreads();   // operand reads done; reuse smem as S^T bounce

  __bf16* st = (__bf16*)smem;   // S^T [d][f], 128x128 (ld 136) = 34816 B
#pragma unroll
  for (int nt = 0; nt < 8; ++nt) {
#pragma unroll
    for (int r = 0; r < 8; ++r) {
      const int f = mt * 16 + 8 * hm + r;
      const int d = nt * 16 + nlo;
      st[d * 136 + f] = (__bf16)acc[nt][r];
    }
  }
  __syncthreads();

  __bf16* sout = Sp + ((size_t)h * NCK + ci) * (size_t)F2 * DD;
  for (int idx = tid; idx < F2 * DD / 8; idx += 256) {    // 2048
    int d = idx >> 4, f8 = (idx & 15) * 8;
    *(v8bf*)(sout + (size_t)d * F2 + f8) = *(const v8bf*)(st + d * 136 + f8);
  }
  if (tid < F2) Zp[((size_t)h * NCK + ci) * F2 + tid] = zval;
}

// ---------------------------------------------------------------------------
// K3: per-head exclusive-by-2 prefix scan of S_j / z_j (layout-agnostic).
//     S_cum[i] = sum_{j <= i-2} S_j.  grid (H), 256 threads.
// ---------------------------------------------------------------------------
__global__ void state_scan_kernel(const __bf16* __restrict__ Sp,
                                  const float* __restrict__ Zp,
                                  __bf16* __restrict__ Sc,
                                  float* __restrict__ Zc) {
  const int h = blockIdx.x, tid = threadIdx.x;
  float acc[64];
#pragma unroll
  for (int j = 0; j < 64; ++j) acc[j] = 0.f;
  float zacc = 0.f;

  for (int i = 0; i < NCK; ++i) {
    const size_t base = ((size_t)h * NCK + i) * (size_t)F2 * DD;
    __bf16* out = Sc + base;
#pragma unroll
    for (int j = 0; j < 8; ++j) {
      v8bf o;
#pragma unroll
      for (int e = 0; e < 8; ++e) o[e] = (__bf16)acc[8 * j + e];
      *(v8bf*)(out + (size_t)j * 2048 + tid * 8) = o;
    }
    if (tid < F2) Zc[((size_t)h * NCK + i) * F2 + tid] = zacc;
    if (i >= 1) {
      const size_t pbase = ((size_t)h * NCK + (i - 1)) * (size_t)F2 * DD;
#pragma unroll
      for (int j = 0; j < 8; ++j) {
        v8bf pv = *(const v8bf*)(Sp + pbase + (size_t)j * 2048 + tid * 8);
#pragma unroll
        for (int e = 0; e < 8; ++e) acc[8 * j + e] += (float)pv[e];
      }
      if (tid < F2) zacc += Zp[((size_t)h * NCK + (i - 1)) * F2 + tid];
    }
  }
}

// ---------------------------------------------------------------------------
// K4: output. grid (H, NC), 256 threads.
//   scores = q @ K_win^T * D^-0.5 (keys = chunks i-1, i), band+causal mask,
//   P = wf * exp(scores - rowmax);  y = P @ V_win + f_q @ S_cum;
//   y /= (rowsum(P) + f_q . z_cum)
// LDS: region0 (64x136 bf16: q -> P -> f_q), region1 (128x136 bf16:
// k row-major -> v^T -> S^T), pm (64x32 f32). All B operands K-contiguous.
// ---------------------------------------------------------------------------
__global__ void output_kernel(const __bf16* __restrict__ Qb,  // [H][L][D]
                              const __bf16* __restrict__ Kb,  // [KVH][L][D]
                              const __bf16* __restrict__ Vb,  // [KVH][L][D]
                              const __bf16* __restrict__ Fq,  // [H][L][F2]
                              const __bf16* __restrict__ Sc,  // [H][NC][D][F2] (S^T)
                              const float*  __restrict__ Zc,  // [H][NC][F2]
                              const float*  __restrict__ WFac,// [H]
                              float* __restrict__ Out) {      // [H][L][D]
  const int h = blockIdx.x, ci = blockIdx.y;
  const int tid = threadIdx.x, wave = tid >> 5, lane = tid & 31;
  const int mt = wave & 3, ntb = (wave >> 2) * 4;
  const int hm = lane >> 4, nlo = lane & 15;

  __shared__ __align__(16) unsigned char smem[60416];
  __shared__ float s_rowmx[64], s_sum[64], s_den[64], s_zc[128];
  __bf16* r0 = (__bf16*)smem;               // 64 x 136 bf16
  __bf16* r1 = (__bf16*)(smem + 17408);     // 128 x 136 bf16
  float*  pm = (float*)(smem + 52224);      // 64 x 32 f32

  const float wf = 1.f / (1.f + __expf(-WFac[h]));
  const float scale = 0.088388347648318447f;  // 128^-0.5
  const v8bf zv = bzero8();
  const int g0 = (ci - 1) * CHK;              // first key of window

  // -------- Phase A: q [m][d] and k row-major [n][d] --------
  const __bf16* qp = Qb + ((size_t)h * LL + (size_t)ci * CHK) * DD;
  for (int idx = tid; idx < CHK * DD / 8; idx += 256) {     // 1024
    int r = idx >> 4, c8 = (idx & 15) * 8;
    *(v8bf*)(r0 + r * 136 + c8) = *(const v8bf*)(qp + (size_t)r * DD + c8);
  }
  const __bf16* kp = Kb + (size_t)(h >> 2) * LL * DD;
  for (int idx = tid; idx < 2 * CHK * DD / 8; idx += 256) { // 2048
    int n = idx >> 4, d8 = (idx & 15) * 8;
    int kg = g0 + n;
    v8bf kv = (kg >= 0) ? *(const v8bf*)(kp + (size_t)kg * DD + d8) : zv;
    *(v8bf*)(r1 + n * 136 + d8) = kv;
  }
  __syncthreads();

  v8f acc[4];
#pragma unroll
  for (int j = 0; j < 4; ++j) acc[j] = vzero8();
#pragma unroll
  for (int kk = 0; kk < 4; ++kk) {
    v16bf a = load_kmaj_tile(r0, 136, mt * 16, kk * 32);
#pragma unroll
    for (int j = 0; j < 4; ++j) {
      v16bf b = load_kmaj_tile(r1, 136, (ntb + j) * 16, kk * 32);
      acc[j] = wmma_bf16(a, b, acc[j]);
    }
  }

  // scale + mask + per-row partial max
#pragma unroll
  for (int r = 0; r < 8; ++r) {
    const int m = mt * 16 + 8 * hm + r;
    float pmax = -3.0e8f;
#pragma unroll
    for (int j = 0; j < 4; ++j) {
      const int c = (ntb + j) * 16 + nlo;
      const bool ok = (c <= m + 64) && ((ci > 0) || (c >= 64));
      const float s = ok ? acc[j][r] * scale : -1.0e8f;
      acc[j][r] = s;
      pmax = fmaxf(pmax, s);
    }
    pm[m * 32 + (wave >> 2) * 16 + nlo] = pmax;
  }
  __syncthreads();     // also: all q / k reads are done
  if (tid < 64) {
    float mx = -3.0e8f;
#pragma unroll 8
    for (int s = 0; s < 32; ++s) mx = fmaxf(mx, pm[tid * 32 + s]);
    s_rowmx[tid] = mx;
  }
  __syncthreads();

  // exp -> probs (bf16 into r0), partial sums; load v^T into r1
  __builtin_prefetch(Sc + ((size_t)h * NCK + ci) * (size_t)F2 * DD, 0, 1);
#pragma unroll
  for (int r = 0; r < 8; ++r) {
    const int m = mt * 16 + 8 * hm + r;
    const float mx = s_rowmx[m];
    float psum = 0.f;
#pragma unroll
    for (int j = 0; j < 4; ++j) {
      const int c = (ntb + j) * 16 + nlo;
      const float p = wf * __expf(acc[j][r] - mx);
      psum += p;
      r0[m * 136 + c] = (__bf16)p;
    }
    pm[m * 32 + (wave >> 2) * 16 + nlo] = psum;
  }
  const __bf16* vp = Vb + (size_t)(h >> 2) * LL * DD;
  for (int idx = tid; idx < 2 * CHK * DD / 8; idx += 256) { // 2048
    int n = idx >> 4, d8 = (idx & 15) * 8;
    int kg = g0 + n;
    v8bf vv = (kg >= 0) ? *(const v8bf*)(vp + (size_t)kg * DD + d8) : zv;
#pragma unroll
    for (int e = 0; e < 8; ++e) r1[(d8 + e) * 136 + n] = vv[e];
  }
  __syncthreads();
  if (tid < 64) {
    float s = 0.f;
#pragma unroll 8
    for (int t = 0; t < 32; ++t) s += pm[tid * 32 + t];
    s_sum[tid] = s;
  }

  // -------- Phase B: y = P(64x128) @ V(128x128), B = v^T [d][n] --------
  v8f y[4];
#pragma unroll
  for (int j = 0; j < 4; ++j) y[j] = vzero8();
#pragma unroll
  for (int kk = 0; kk < 4; ++kk) {
    v16bf a = load_kmaj_tile(r0, 136, mt * 16, kk * 32);
#pragma unroll
    for (int j = 0; j < 4; ++j) {
      v16bf b = load_kmaj_tile(r1, 136, (ntb + j) * 16, kk * 32);
      y[j] = wmma_bf16(a, b, y[j]);
    }
  }
  __syncthreads();     // all P / v^T reads done, regions reusable

  // -------- Phase C: y += f_q(64x128) @ S_cum, B = S^T [d][f] --------
  const __bf16* fqp = Fq + ((size_t)h * LL + (size_t)ci * CHK) * F2;
  for (int idx = tid; idx < CHK * F2 / 8; idx += 256) {     // 1024
    int r = idx >> 4, c8 = (idx & 15) * 8;
    *(v8bf*)(r0 + r * 136 + c8) = *(const v8bf*)(fqp + (size_t)r * F2 + c8);
  }
  const __bf16* scp = Sc + ((size_t)h * NCK + ci) * (size_t)F2 * DD;
  for (int idx = tid; idx < F2 * DD / 8; idx += 256) {      // 2048
    int d = idx >> 4, f8 = (idx & 15) * 8;
    *(v8bf*)(r1 + d * 136 + f8) = *(const v8bf*)(scp + (size_t)d * F2 + f8);
  }
  if (tid < F2) s_zc[tid] = Zc[((size_t)h * NCK + ci) * F2 + tid];
  __syncthreads();

#pragma unroll
  for (int kk = 0; kk < 4; ++kk) {
    v16bf a = load_kmaj_tile(r0, 136, mt * 16, kk * 32);
#pragma unroll
    for (int j = 0; j < 4; ++j) {
      v16bf b = load_kmaj_tile(r1, 136, (ntb + j) * 16, kk * 32);
      y[j] = wmma_bf16(a, b, y[j]);
    }
  }

  // denominator: sum_sm + f_q . z_cum
  if (tid < 64) {
    float dsum = 0.f;
#pragma unroll 8
    for (int f = 0; f < F2; ++f) dsum += (float)r0[tid * 136 + f] * s_zc[f];
    s_den[tid] = s_sum[tid] + dsum;
  }
  __syncthreads();

  float* op = Out + ((size_t)h * LL + (size_t)ci * CHK) * DD;
#pragma unroll
  for (int r = 0; r < 8; ++r) {
    const int m = mt * 16 + 8 * hm + r;
    const float inv = 1.f / s_den[m];
#pragma unroll
    for (int j = 0; j < 4; ++j) {
      const int c = (ntb + j) * 16 + nlo;
      op[(size_t)m * DD + c] = y[j][r] * inv;
    }
  }
}

// ---------------------------------------------------------------------------
// launch
// ---------------------------------------------------------------------------
extern "C" void kernel_launch(void* const* d_in, const int* in_sizes, int n_in,
                              void* d_out, int out_size, void* d_ws,
                              size_t ws_size, hipStream_t stream) {
  (void)in_sizes; (void)n_in; (void)out_size; (void)ws_size;

  const float* q    = (const float*)d_in[0];  // [1,32,2048,128]
  const float* k    = (const float*)d_in[1];  // [1, 8,2048,128]
  const float* v    = (const float*)d_in[2];  // [1, 8,2048,128]
  const float* Wq   = (const float*)d_in[3];  // [32,128,64]
  const float* Wk   = (const float*)d_in[4];  // [32,128,64]
  const float* wfac = (const float*)d_in[5];  // [1,32,1,1]
  float* out = (float*)d_out;                 // [1,32,2048,128] f32

  unsigned char* ws = (unsigned char*)d_ws;
  size_t off = 0;
  __bf16* qbf = (__bf16*)(ws + off); off += (size_t)HH * LL * DD * 2;    // 16 MB
  __bf16* kbf = (__bf16*)(ws + off); off += (size_t)KVHH * LL * DD * 2;  //  4 MB
  __bf16* vbf = (__bf16*)(ws + off); off += (size_t)KVHH * LL * DD * 2;  //  4 MB
  __bf16* fq  = (__bf16*)(ws + off); off += (size_t)HH * LL * F2 * 2;    // 16 MB
  __bf16* fk  = (__bf16*)(ws + off); off += (size_t)HH * LL * F2 * 2;    // 16 MB
  __bf16* Sp  = (__bf16*)(ws + off); off += (size_t)HH * NCK * F2 * DD * 2; // 32 MB
  __bf16* Sc  = (__bf16*)(ws + off); off += (size_t)HH * NCK * F2 * DD * 2; // 32 MB
  float*  Zp  = (float*)(ws + off);  off += (size_t)HH * NCK * F2 * 4;   // 512 KB
  float*  Zc  = (float*)(ws + off);  off += (size_t)HH * NCK * F2 * 4;   // 512 KB

  const int nq4 = HH * LL * DD / 4, nkv4 = KVHH * LL * DD / 4;
  cvt_f32_bf16_kernel<<<(nq4 + 255) / 256, 256, 0, stream>>>(q, qbf, nq4);
  cvt_f32_bf16_kernel<<<(nkv4 + 255) / 256, 256, 0, stream>>>(k, kbf, nkv4);
  cvt_f32_bf16_kernel<<<(nkv4 + 255) / 256, 256, 0, stream>>>(v, vbf, nkv4);

  feature_kernel<<<dim3(HH, NCK), 256, 0, stream>>>(q, Wq, fq, 1);
  feature_kernel<<<dim3(HH, NCK), 256, 0, stream>>>(k, Wk, fk, 4);

  state_partial_kernel<<<dim3(HH, NCK), 256, 0, stream>>>(fk, vbf, Sp, Zp);
  state_scan_kernel<<<HH, 256, 0, stream>>>(Sp, Zp, Sc, Zc);

  output_kernel<<<dim3(HH, NCK), 256, 0, stream>>>(qbf, kbf, vbf, fq, Sc, Zc,
                                                   wfac, out);
}
